// GeometricEncoder_85899345943
// MI455X (gfx1250) — matrix operand order repaired and checked by reading, not compiled, not run on hardware
//
#include <hip/hip_runtime.h>
#include <hip/hip_bf16.h>

// ---------------------------------------------------------------------------
// GeometricEncoder for MI455X (gfx1250): bf16 WMMA GEMMs + fused elementwise.
// B=32 NK=196(->224 padded) D_IN=2048 D_MODEL=512 H=8 DK=DV=64 DFF=2048 L=3
// ---------------------------------------------------------------------------

#define B_    32
#define NK_   196
#define NKP_  224      // padded to multiple of 32 for WMMA K / 16 for tiles
#define DIN_  2048
#define DM_   512
#define H_    8
#define DK_   64
#define DFF_  2048
#define NL_   3

typedef __bf16 bf16_t;
typedef __attribute__((ext_vector_type(16))) __bf16 v16bf;
typedef __attribute__((ext_vector_type(8)))  __bf16 v8bf;
typedef __attribute__((ext_vector_type(8)))  float  v8f;

// Global-address-space pointers: force global_load_b128 (not flat_load).
typedef const __attribute__((address_space(1))) v8bf*  gv8bf_p;
typedef const __attribute__((address_space(1))) v16bf* gv16bf_p;

static __device__ inline v16bf pack16(v8bf lo, v8bf hi) {
  v16bf r;
#pragma unroll
  for (int i = 0; i < 8; i++) { r[i] = lo[i]; r[i + 8] = hi[i]; }
  return r;
}

// A fragment (16-bit A 16x32 ISA layout): lane L<16 holds row L, K {0..7} and
// {16..23}; lane L>=16 holds row L-16, K {8..15} and {24..31}.  Two b128
// global loads at byte offsets 0 and 32.  base uniform, off per-lane (int).
static __device__ inline v16bf load_a_g(const bf16_t* base, int off) {
  return pack16(*(gv8bf_p)(base + off), *(gv8bf_p)(base + off + 16));
}
// B fragment from W^T[N][K]: lane n<16 -> K {0..7} and {16..23} of row n,
// n>=16 -> K {8..15} and {24..31}.  Two b128 global loads.
static __device__ inline v16bf load_b_g(const bf16_t* base, int off) {
  return pack16(*(gv8bf_p)(base + off), *(gv8bf_p)(base + off + 16));
}

// ---------------------------------------------------------------------------
// Batched GEMM: C[M,N] = act(A[M,K] * Bt[N,K]^T + bias)
// One wave computes a 32 x (NT*16) tile.  K loop manually unrolled by 2 with
// ping-pong fragment sets; uniform bases advance (scalar adds) while per-lane
// offsets stay constant 32-bit ints -> SADDR+voffset global loads.
// Grid: x = M/32, y = N-tile group, z = batch.
// ---------------------------------------------------------------------------
template <int NT>
__global__ void __launch_bounds__(32)
gemm_wmma_bf16(const bf16_t* __restrict__ A, long long sA, int lda,
               const bf16_t* __restrict__ Bt, long long sB, int ldb,
               const float* __restrict__ bias,
               float* __restrict__ C, long long sC, int ldc,
               bf16_t* __restrict__ Cbf,
               int K, int relu, int nt_base) {
  const int lane = threadIdx.x & 31;
  const int m0 = blockIdx.x * 32;
  const int n0 = (nt_base + blockIdx.y * NT) * 16;

  const bf16_t* pA = A + (size_t)blockIdx.z * sA;   // uniform base
  const bf16_t* pB = Bt + (size_t)blockIdx.z * sB;  // uniform base
  float*        Cb = C + (size_t)blockIdx.z * sC;

  const int arow = lane & 15;
  const int aoff0 = (m0 + arow) * lda + (lane >> 4) * 8;   // per-lane, const
  const int aoff1 = aoff0 + 16 * lda;
  int boffs[NT];
#pragma unroll
  for (int t = 0; t < NT; t++)
    boffs[t] = (n0 + t * 16 + arow) * ldb + (lane >> 4) * 8;

  v8f acc0[NT], acc1[NT];
#pragma unroll
  for (int t = 0; t < NT; t++)
#pragma unroll
    for (int r = 0; r < 8; r++) { acc0[t][r] = 0.f; acc1[t][r] = 0.f; }

  const int steps = K >> 5;  // number of K=32 chunks, always >= 2 here

  // chunk 0 -> set A
  v16bf a0A = load_a_g(pA, aoff0);
  v16bf a1A = load_a_g(pA, aoff1);
  v16bf bA[NT];
#pragma unroll
  for (int t = 0; t < NT; t++) bA[t] = load_b_g(pB, boffs[t]);
  pA += 32;  // scalar base advance to chunk 1
  pB += 32;

  v16bf a0B, a1B, bB[NT];
  int s = 1;
  // invariant at loop top: set A holds chunk s-1 (uncomputed); bases at s.
  for (; s + 1 < steps; s += 2) {
    // load chunk s into set B
    a0B = load_a_g(pA, aoff0);
    a1B = load_a_g(pA, aoff1);
#pragma unroll
    for (int t = 0; t < NT; t++) bB[t] = load_b_g(pB, boffs[t]);
    // compute chunk s-1 (set A)
#pragma unroll
    for (int t = 0; t < NT; t++)
      acc0[t] = __builtin_amdgcn_wmma_f32_16x16x32_bf16(
          false, a0A, false, bA[t], (short)0, acc0[t], false, false);
#pragma unroll
    for (int t = 0; t < NT; t++)
      acc1[t] = __builtin_amdgcn_wmma_f32_16x16x32_bf16(
          false, a1A, false, bA[t], (short)0, acc1[t], false, false);
    // load chunk s+1 into set A (+32 elements folds into load offset)
    a0A = load_a_g(pA, aoff0 + 32);
    a1A = load_a_g(pA, aoff1 + 32);
#pragma unroll
    for (int t = 0; t < NT; t++) bA[t] = load_b_g(pB, boffs[t] + 32);
    // compute chunk s (set B)
#pragma unroll
    for (int t = 0; t < NT; t++)
      acc0[t] = __builtin_amdgcn_wmma_f32_16x16x32_bf16(
          false, a0B, false, bB[t], (short)0, acc0[t], false, false);
#pragma unroll
    for (int t = 0; t < NT; t++)
      acc1[t] = __builtin_amdgcn_wmma_f32_16x16x32_bf16(
          false, a1B, false, bB[t], (short)0, acc1[t], false, false);
    pA += 64;  // scalar adds only
    pB += 64;
  }

  if (s < steps) {
    // one trailing chunk s: load into B, compute A then B
    a0B = load_a_g(pA, aoff0);
    a1B = load_a_g(pA, aoff1);
#pragma unroll
    for (int t = 0; t < NT; t++) bB[t] = load_b_g(pB, boffs[t]);
#pragma unroll
    for (int t = 0; t < NT; t++)
      acc0[t] = __builtin_amdgcn_wmma_f32_16x16x32_bf16(
          false, a0A, false, bA[t], (short)0, acc0[t], false, false);
#pragma unroll
    for (int t = 0; t < NT; t++)
      acc1[t] = __builtin_amdgcn_wmma_f32_16x16x32_bf16(
          false, a1A, false, bA[t], (short)0, acc1[t], false, false);
#pragma unroll
    for (int t = 0; t < NT; t++)
      acc0[t] = __builtin_amdgcn_wmma_f32_16x16x32_bf16(
          false, a0B, false, bB[t], (short)0, acc0[t], false, false);
#pragma unroll
    for (int t = 0; t < NT; t++)
      acc1[t] = __builtin_amdgcn_wmma_f32_16x16x32_bf16(
          false, a1B, false, bB[t], (short)0, acc1[t], false, false);
  } else {
    // only set A pending
#pragma unroll
    for (int t = 0; t < NT; t++)
      acc0[t] = __builtin_amdgcn_wmma_f32_16x16x32_bf16(
          false, a0A, false, bA[t], (short)0, acc0[t], false, false);
#pragma unroll
    for (int t = 0; t < NT; t++)
      acc1[t] = __builtin_amdgcn_wmma_f32_16x16x32_bf16(
          false, a1A, false, bA[t], (short)0, acc1[t], false, false);
  }

  // C/D layout: VGPR r -> row r + 8*(lane>=16), col = lane&15
  const int col = lane & 15;
  const int rb  = (lane >> 4) * 8;
#pragma unroll
  for (int t = 0; t < NT; t++) {
    int n = n0 + t * 16 + col;
    float bv = bias ? bias[n] : 0.f;
#pragma unroll
    for (int r = 0; r < 8; r++) {
      {
        int m = m0 + rb + r;
        float v = acc0[t][r] + bv;
        if (relu) v = v > 0.f ? v : 0.f;
        size_t idx = (size_t)m * ldc + n;
        Cb[idx] = v;
        if (Cbf) Cbf[(size_t)blockIdx.z * sC + idx] = (bf16_t)v;
      }
      {
        int m = m0 + 16 + rb + r;
        float v = acc1[t][r] + bv;
        if (relu) v = v > 0.f ? v : 0.f;
        size_t idx = (size_t)m * ldc + n;
        Cb[idx] = v;
        if (Cbf) Cbf[(size_t)blockIdx.z * sC + idx] = (bf16_t)v;
      }
    }
  }
}

// ---------------------------------------------------------------------------
// Input: bf16 convert + padding mask (row feature sum == 0)
// ---------------------------------------------------------------------------
__global__ void __launch_bounds__(256)
prep_input(const float* __restrict__ in, bf16_t* __restrict__ out,
           int* __restrict__ maskb) {
  int row = blockIdx.x;
  const float* r = in + (size_t)row * DIN_;
  bf16_t* o = out + (size_t)row * DIN_;
  float s = 0.f;
  for (int i = threadIdx.x; i < DIN_; i += 256) {
    float v = r[i];
    s += v;
    o[i] = (bf16_t)v;
  }
  __shared__ float red[256];
  red[threadIdx.x] = s;
  __syncthreads();
  for (int ofs = 128; ofs > 0; ofs >>= 1) {
    if (threadIdx.x < ofs) red[threadIdx.x] += red[threadIdx.x + ofs];
    __syncthreads();
  }
  if (threadIdx.x == 0) maskb[row] = (red[0] == 0.f) ? 1 : 0;
}

// W[K][N] fp32 -> Wt[N][K] bf16
__global__ void __launch_bounds__(256)
transpose_bf16(const float* __restrict__ W, bf16_t* __restrict__ Wt, int K,
               int N) {
  int idx = blockIdx.x * blockDim.x + threadIdx.x;
  if (idx >= K * N) return;
  int k = idx / N, n = idx % N;
  Wt[(size_t)n * K + k] = (bf16_t)W[idx];
}

// ---------------------------------------------------------------------------
// log_wg[b,h,q,k] = log(clip(relu(emb(b,q,k,:)·Wg[h,:] + bg[h]), 1e-6))
// emb computed on the fly from boxes (sin/cos of 32 scaled log-deltas).
// ---------------------------------------------------------------------------
__global__ void __launch_bounds__(128)
geom_bias_kernel(const float* __restrict__ boxes, const float* __restrict__ Wg,
                 const float* __restrict__ bg, float* __restrict__ logwg) {
  __shared__ float sWg[H_ * 64];
  __shared__ float sbg[H_];
  for (int i = threadIdx.x; i < H_ * 64; i += blockDim.x) sWg[i] = Wg[i];
  if (threadIdx.x < H_) sbg[threadIdx.x] = bg[threadIdx.x];
  __syncthreads();

  long long idx = (long long)blockIdx.x * blockDim.x + threadIdx.x;
  const long long total = (long long)B_ * NK_ * NK_;
  if (idx >= total) return;
  int k = (int)(idx % NK_);
  long long t = idx / NK_;
  int q = (int)(t % NK_);
  int b = (int)(t / NK_);

  const float* pq = boxes + ((size_t)b * NK_ + q) * 4;
  const float* pk = boxes + ((size_t)b * NK_ + k) * 4;
  float cxq = (pq[0] + pq[2]) * 0.5f, cyq = (pq[1] + pq[3]) * 0.5f;
  float wq = (pq[2] - pq[0]) + 1.f,   hq  = (pq[3] - pq[1]) + 1.f;
  float cxk = (pk[0] + pk[2]) * 0.5f, cyk = (pk[1] + pk[3]) * 0.5f;
  float wk = (pk[2] - pk[0]) + 1.f,   hk  = (pk[3] - pk[1]) + 1.f;

  float pos[4];
  pos[0] = __logf(fmaxf(fabsf((cxq - cxk) / wq), 1e-3f));
  pos[1] = __logf(fmaxf(fabsf((cyq - cyk) / hq), 1e-3f));
  pos[2] = __logf(wq / wk);
  pos[3] = __logf(hq / hk);

  float acc[H_];
#pragma unroll
  for (int h = 0; h < H_; h++) acc[h] = sbg[h];

#pragma unroll
  for (int p = 0; p < 4; p++) {
#pragma unroll
    for (int f = 0; f < 8; f++) {
      float invf = __expf(-6.90775527898f * (0.125f * (float)f));  // 1000^(-f/8)
      float m = 100.f * pos[p] * invf;
      float sn, cs;
      __sincosf(m, &sn, &cs);
      int gs = p * 8 + f, gc = 32 + p * 8 + f;
#pragma unroll
      for (int h = 0; h < H_; h++)
        acc[h] += sWg[h * 64 + gs] * sn + sWg[h * 64 + gc] * cs;
    }
  }
#pragma unroll
  for (int h = 0; h < H_; h++) {
    float w = fmaxf(fmaxf(acc[h], 0.f), 1e-6f);
    logwg[(((size_t)(b * H_ + h)) * NK_ + q) * NK_ + k] = __logf(w);
  }
}

// ---------------------------------------------------------------------------
// out = LN(xin (+ res)); writes fp32 and bf16 copies.  1 block per 512 row.
// ---------------------------------------------------------------------------
__global__ void __launch_bounds__(256)
add_ln(const float* __restrict__ xin, const float* __restrict__ res,
       const float* __restrict__ g, const float* __restrict__ beta,
       float* __restrict__ outf, bf16_t* __restrict__ outb) {
  int row = blockIdx.x;
  int t = threadIdx.x;
  size_t base = (size_t)row * DM_;
  float e0 = xin[base + t], e1 = xin[base + t + 256];
  if (res) { e0 += res[base + t]; e1 += res[base + t + 256]; }
  __shared__ float rs[256], rq[256];
  rs[t] = e0 + e1;
  rq[t] = e0 * e0 + e1 * e1;
  __syncthreads();
  for (int o = 128; o > 0; o >>= 1) {
    if (t < o) { rs[t] += rs[t + o]; rq[t] += rq[t + o]; }
    __syncthreads();
  }
  float mean = rs[0] * (1.f / DM_);
  float var = rq[0] * (1.f / DM_) - mean * mean;
  float inv = rsqrtf(var + 1e-5f);
  float o0 = (e0 - mean) * inv * g[t] + beta[t];
  float o1 = (e1 - mean) * inv * g[t + 256] + beta[t + 256];
  outf[base + t] = o0;
  outf[base + t + 256] = o1;
  outb[base + t] = (bf16_t)o0;
  outb[base + t + 256] = (bf16_t)o1;
}

// qkv fp32 [B*NK, 512] -> per-head bf16, zero-padded to NKP rows.
// vmode=0: out[b,h,q,d]  (Q/K);  vmode=1: out[b,h,d,q]  (V transposed)
__global__ void __launch_bounds__(256)
pack_heads(const float* __restrict__ in, bf16_t* __restrict__ out, int vmode) {
  int idx = blockIdx.x * blockDim.x + threadIdx.x;
  const int total = B_ * H_ * NKP_ * DK_;
  if (idx >= total) return;
  int d = idx & (DK_ - 1);
  int t = idx / DK_;
  int q = t % NKP_; t /= NKP_;
  int h = t % H_;
  int b = t / H_;
  float v = (q < NK_) ? in[((size_t)b * NK_ + q) * DM_ + h * DK_ + d] : 0.f;
  if (vmode)
    out[(((size_t)(b * H_ + h)) * DK_ + d) * NKP_ + q] = (bf16_t)v;
  else
    out[(((size_t)(b * H_ + h)) * NKP_ + q) * DK_ + d] = (bf16_t)v;
}

// softmax over keys: logits = mask ? -1e9 : log_wg + scale*scores
__global__ void __launch_bounds__(256)
softmax_probs(const float* __restrict__ scores, const float* __restrict__ logwg,
              const int* __restrict__ maskb, bf16_t* __restrict__ probs,
              float scale) {
  int q = blockIdx.x % NKP_;
  int bh = blockIdx.x / NKP_;
  int b = bh / H_;
  bf16_t* pr = probs + ((size_t)bh * NKP_ + q) * NKP_;
  int k = threadIdx.x;
  if (q >= NK_) {
    if (k < NKP_) pr[k] = (bf16_t)0.f;
    return;
  }
  float logit = -3.0e38f;
  if (k < NK_) {
    float s = scores[((size_t)bh * NKP_ + q) * NKP_ + k];
    float lw = logwg[((size_t)bh * NK_ + q) * NK_ + k];
    logit = maskb[b * NK_ + k] ? -1e9f : (lw + s * scale);
  }
  __shared__ float red[256];
  red[k] = logit;
  __syncthreads();
  for (int o = 128; o > 0; o >>= 1) {
    if (k < o) red[k] = fmaxf(red[k], red[k + o]);
    __syncthreads();
  }
  float mx = red[0];
  __syncthreads();
  float e = (k < NK_) ? __expf(logit - mx) : 0.f;
  red[k] = e;
  __syncthreads();
  for (int o = 128; o > 0; o >>= 1) {
    if (k < o) red[k] += red[k + o];
    __syncthreads();
  }
  float inv = 1.f / red[0];
  if (k < NKP_) pr[k] = (bf16_t)(e * inv);
}

// o fp32 [B,H,NKP,DK] -> bf16 [B*NK, H*DK] (drop padded rows)
__global__ void __launch_bounds__(256)
pack_o_kernel(const float* __restrict__ o, bf16_t* __restrict__ out) {
  int idx = blockIdx.x * blockDim.x + threadIdx.x;
  const int total = B_ * NK_ * DM_;
  if (idx >= total) return;
  int c = idx % DM_;
  int r = idx / DM_;
  int q = r % NK_;
  int b = r / NK_;
  int h = c / DK_;
  int d = c % DK_;
  out[idx] = (bf16_t)o[(((size_t)(b * H_ + h)) * NKP_ + q) * DK_ + d];
}

// ---------------------------------------------------------------------------
extern "C" void kernel_launch(void* const* d_in, const int* in_sizes, int n_in,
                              void* d_out, int out_size, void* d_ws,
                              size_t ws_size, hipStream_t stream) {
  (void)in_sizes; (void)n_in; (void)out_size; (void)ws_size;
  const float* input    = (const float*)d_in[0];
  const float* boxes    = (const float*)d_in[1];
  const float* Wfc      = (const float*)d_in[2];
  const float* bfc      = (const float*)d_in[3];
  const float* ln0_g    = (const float*)d_in[4];
  const float* ln0_b    = (const float*)d_in[5];
  const float* Wg       = (const float*)d_in[6];
  const float* bg       = (const float*)d_in[7];
  const float* Wq       = (const float*)d_in[8];
  const float* bq       = (const float*)d_in[9];
  const float* Wk       = (const float*)d_in[10];
  const float* bk       = (const float*)d_in[11];
  const float* Wv       = (const float*)d_in[12];
  const float* bv       = (const float*)d_in[13];
  const float* Wo       = (const float*)d_in[14];
  const float* bo       = (const float*)d_in[15];
  const float* ln_att_g = (const float*)d_in[16];
  const float* ln_att_b = (const float*)d_in[17];
  const float* W1       = (const float*)d_in[18];
  const float* b1       = (const float*)d_in[19];
  const float* W2       = (const float*)d_in[20];
  const float* b2       = (const float*)d_in[21];
  const float* ln_ff_g  = (const float*)d_in[22];
  const float* ln_ff_b  = (const float*)d_in[23];

  char* ws = (char*)d_ws;
  size_t off = 0;
  auto alloc = [&](size_t bytes) -> void* {
    void* p = ws + off;
    off += (bytes + 255) & ~(size_t)255;
    return p;
  };
  const int MROWS = B_ * NK_;  // 6272
  bf16_t* in_bf = (bf16_t*)alloc((size_t)MROWS * DIN_ * 2);
  int*    maskb = (int*)alloc((size_t)MROWS * 4);
  bf16_t* WfcT  = (bf16_t*)alloc((size_t)DM_ * DIN_ * 2);
  bf16_t* WqT   = (bf16_t*)alloc((size_t)NL_ * DM_ * DM_ * 2);
  bf16_t* WkT   = (bf16_t*)alloc((size_t)NL_ * DM_ * DM_ * 2);
  bf16_t* WvT   = (bf16_t*)alloc((size_t)NL_ * DM_ * DM_ * 2);
  bf16_t* WoT   = (bf16_t*)alloc((size_t)NL_ * DM_ * DM_ * 2);
  bf16_t* W1T   = (bf16_t*)alloc((size_t)NL_ * DFF_ * DM_ * 2);
  bf16_t* W2T   = (bf16_t*)alloc((size_t)NL_ * DM_ * DFF_ * 2);
  float*  logwg = (float*)alloc((size_t)B_ * H_ * NK_ * NK_ * 4);
  float*  xf    = (float*)alloc((size_t)MROWS * DM_ * 4);
  bf16_t* xbf   = (bf16_t*)alloc((size_t)MROWS * DM_ * 2);
  float*  tmp   = (float*)alloc((size_t)MROWS * DFF_ * 4);
  bf16_t* hbf   = (bf16_t*)alloc((size_t)MROWS * DFF_ * 2);
  bf16_t* Qh    = (bf16_t*)alloc((size_t)B_ * H_ * NKP_ * DK_ * 2);
  bf16_t* Kh    = (bf16_t*)alloc((size_t)B_ * H_ * NKP_ * DK_ * 2);
  bf16_t* VhT   = (bf16_t*)alloc((size_t)B_ * H_ * DK_ * NKP_ * 2);
  float*  scr   = (float*)alloc((size_t)B_ * H_ * NKP_ * NKP_ * 4);
  bf16_t* probs = (bf16_t*)alloc((size_t)B_ * H_ * NKP_ * NKP_ * 2);
  float*  obuf  = (float*)alloc((size_t)B_ * H_ * NKP_ * DK_ * 4);
  bf16_t* opk   = (bf16_t*)alloc((size_t)MROWS * DM_ * 2);

  // Generic GEMM dispatch: NT=4 bulk + NT={1,2,3} remainder launch.
  auto launch_gemm = [&](const bf16_t* A, long long sA, int lda,
                         const bf16_t* Bt, long long sB, int ldb,
                         const float* bias, float* C, long long sC, int ldc,
                         bf16_t* Cbf, int M, int N, int K, int relu,
                         int batches) {
    int mtiles = M / 32;
    int groups = N / 64;
    int rem = (N % 64) / 16;
    if (groups)
      gemm_wmma_bf16<4><<<dim3(mtiles, groups, batches), 32, 0, stream>>>(
          A, sA, lda, Bt, sB, ldb, bias, C, sC, ldc, Cbf, K, relu, 0);
    if (rem == 1)
      gemm_wmma_bf16<1><<<dim3(mtiles, 1, batches), 32, 0, stream>>>(
          A, sA, lda, Bt, sB, ldb, bias, C, sC, ldc, Cbf, K, relu, groups * 4);
    else if (rem == 2)
      gemm_wmma_bf16<2><<<dim3(mtiles, 1, batches), 32, 0, stream>>>(
          A, sA, lda, Bt, sB, ldb, bias, C, sC, ldc, Cbf, K, relu, groups * 4);
    else if (rem == 3)
      gemm_wmma_bf16<3><<<dim3(mtiles, 1, batches), 32, 0, stream>>>(
          A, sA, lda, Bt, sB, ldb, bias, C, sC, ldc, Cbf, K, relu, groups * 4);
  };

  // --- precompute ---
  prep_input<<<MROWS, 256, 0, stream>>>(input, in_bf, maskb);
  transpose_bf16<<<(DIN_ * DM_ + 255) / 256, 256, 0, stream>>>(Wfc, WfcT, DIN_, DM_);
  for (int i = 0; i < NL_; i++) {
    size_t so = (size_t)i * DM_ * DM_;
    transpose_bf16<<<(DM_ * DM_ + 255) / 256, 256, 0, stream>>>(Wq + so, WqT + so, DM_, DM_);
    transpose_bf16<<<(DM_ * DM_ + 255) / 256, 256, 0, stream>>>(Wk + so, WkT + so, DM_, DM_);
    transpose_bf16<<<(DM_ * DM_ + 255) / 256, 256, 0, stream>>>(Wv + so, WvT + so, DM_, DM_);
    transpose_bf16<<<(DM_ * DM_ + 255) / 256, 256, 0, stream>>>(Wo + so, WoT + so, DM_, DM_);
    size_t sf = (size_t)i * DM_ * DFF_;
    transpose_bf16<<<(DM_ * DFF_ + 255) / 256, 256, 0, stream>>>(W1 + sf, W1T + sf, DM_, DFF_);
    transpose_bf16<<<(DFF_ * DM_ + 255) / 256, 256, 0, stream>>>(W2 + sf, W2T + sf, DFF_, DM_);
  }
  {
    long long tot = (long long)B_ * NK_ * NK_;
    geom_bias_kernel<<<(unsigned)((tot + 127) / 128), 128, 0, stream>>>(boxes, Wg, bg, logwg);
  }

  // --- x = LN(relu(input @ Wfc + bfc)) ---
  launch_gemm(in_bf, 0, DIN_, WfcT, 0, DIN_, bfc, tmp, 0, DM_, nullptr,
              MROWS, DM_, DIN_, 1, 1);
  add_ln<<<MROWS, 256, 0, stream>>>(tmp, nullptr, ln0_g, ln0_b, xf, xbf);

  const float scale = 0.125f;  // 1/sqrt(64)
  for (int i = 0; i < NL_; i++) {
    size_t so = (size_t)i * DM_ * DM_;
    size_t sf = (size_t)i * DM_ * DFF_;
    const int NPK = B_ * H_ * NKP_ * DK_;
    // Q,K,V projections + head packing (V packed transposed for WMMA B)
    launch_gemm(xbf, 0, DM_, WqT + so, 0, DM_, bq + i * DM_, tmp, 0, DM_,
                nullptr, MROWS, DM_, DM_, 0, 1);
    pack_heads<<<(NPK + 255) / 256, 256, 0, stream>>>(tmp, Qh, 0);
    launch_gemm(xbf, 0, DM_, WkT + so, 0, DM_, bk + i * DM_, tmp, 0, DM_,
                nullptr, MROWS, DM_, DM_, 0, 1);
    pack_heads<<<(NPK + 255) / 256, 256, 0, stream>>>(tmp, Kh, 0);
    launch_gemm(xbf, 0, DM_, WvT + so, 0, DM_, bv + i * DM_, tmp, 0, DM_,
                nullptr, MROWS, DM_, DM_, 0, 1);
    pack_heads<<<(NPK + 255) / 256, 256, 0, stream>>>(tmp, VhT, 1);
    // scores = Q @ K^T (batched over B*H); scale folded into softmax
    launch_gemm(Qh, (long long)NKP_ * DK_, DK_, Kh, (long long)NKP_ * DK_, DK_,
                nullptr, scr, (long long)NKP_ * NKP_, NKP_, nullptr,
                NKP_, NKP_, DK_, 0, B_ * H_);
    softmax_probs<<<B_ * H_ * NKP_, 256, 0, stream>>>(scr, logwg, maskb, probs, scale);
    // o = probs @ V (batched)
    launch_gemm(probs, (long long)NKP_ * NKP_, NKP_, VhT,
                (long long)DK_ * NKP_, NKP_, nullptr, obuf,
                (long long)NKP_ * DK_, DK_, nullptr, NKP_, DK_, NKP_, 0,
                B_ * H_);
    pack_o_kernel<<<(MROWS * DM_ + 255) / 256, 256, 0, stream>>>(obuf, opk);
    // output projection + residual LN
    launch_gemm(opk, 0, DM_, WoT + so, 0, DM_, bo + i * DM_, tmp, 0, DM_,
                nullptr, MROWS, DM_, DM_, 0, 1);
    add_ln<<<MROWS, 256, 0, stream>>>(tmp, xf, ln_att_g + i * DM_,
                                      ln_att_b + i * DM_, xf, xbf);
    // FFN
    launch_gemm(xbf, 0, DM_, W1T + sf, 0, DM_, b1 + i * DFF_, tmp, 0, DFF_,
                hbf, MROWS, DFF_, DM_, 1, 1);
    launch_gemm(hbf, 0, DFF_, W2T + sf, 0, DFF_, b2 + i * DM_, tmp, 0, DM_,
                nullptr, MROWS, DM_, DFF_, 0, 1);
    float* xdst = (i == NL_ - 1) ? (float*)d_out : xf;
    add_ln<<<MROWS, 256, 0, stream>>>(tmp, xf, ln_ff_g + i * DM_,
                                      ln_ff_b + i * DM_, xdst, xbf);
  }
}
